// IJEPAMask2Former_48670569398857
// MI455X (gfx1250) — compile-verified, matrix operand order checked by reading, not run on hardware
//
#include <hip/hip_runtime.h>
#include <hip/hip_bf16.h>
#include <math.h>

#define B_  8
#define Q_  100
#define D_  256
#define H_  8
#define HD_ 32
#define L_  9
#define F_  2048
#define C_  151
#define HM_ 112

typedef __bf16 bf16_t;
typedef bf16_t v16bf __attribute__((ext_vector_type(16)));
typedef bf16_t v8bf  __attribute__((ext_vector_type(8)));
typedef float  v8f   __attribute__((ext_vector_type(8)));

static __device__ __forceinline__ bf16_t f2bf(float f) {
  unsigned u = __builtin_bit_cast(unsigned, f);
  unsigned r = u + 0x7fffu + ((u >> 16) & 1u);      // round-to-nearest-even
  unsigned short s = (unsigned short)(r >> 16);
  return __builtin_bit_cast(bf16_t, s);
}

// ---------------------------------------------------------------------------
// Tiled WMMA GEMM: C[M,N] = act(A[M,K] @ B[K,N] + bias[N])
// f32 in/out, bf16 fragments, f32 accumulate.
// Block tile 64(M) x 64(N); 4 waves in 2x2, each wave computes 32x32 via
// 2 A-fragments x 2 B-fragments -> 4 WMMAs per k-step (k-step = 32).
// K must be a multiple of 32 (true for all calls: 256 or 2048).
//
// LDS tiles are stored in fragment-ready permuted K order
// [k0..7 | k16..23 | k8..15 | k24..31] so a fragment is two ds_load_b128:
// lanes 0-15 read row bytes [0,32), lanes 16-31 read [32,64).
// ---------------------------------------------------------------------------
#define TM  64
#define TN  64
#define LDT 40   // padded LDS row stride in elements (80B, 16B-aligned)

__global__ __launch_bounds__(128)
void gemm_bf16_wmma(const float* __restrict__ A, const float* __restrict__ Bw,
                    float* __restrict__ C, const float* __restrict__ bias,
                    int M, int N, int K, long sA, long sB, long sC, int act) {
  __shared__ __attribute__((aligned(32))) bf16_t As[TM * LDT];  // [m][kperm]
  __shared__ __attribute__((aligned(32))) bf16_t Bs[TN * LDT];  // [n][kperm]
  const int z = blockIdx.z;
  A  += (long)z * sA;
  Bw += (long)z * sB;
  C  += (long)z * sC;
  const int mBase = blockIdx.y * TM;
  const int nBase = blockIdx.x * TN;
  const int tid  = threadIdx.x;
  const int lane = tid & 31;
  const int wave = tid >> 5;
  const int mOff = (wave >> 1) * 32;    // wave's 32-row strip
  const int nOff = (wave &  1) * 32;    // wave's 32-col strip
  // staging roles: each thread owns one row and one 8-wide k-group
  const int srow = tid & 31;            // row/col within a 32-strip
  const int kc   = (tid >> 5) * 8;      // k-chunk base: 0,8,16,24
  const int kpos = (kc == 8) ? 16 : (kc == 16) ? 8 : kc;  // permuted position

  v8f acc00 = {}, acc01 = {}, acc10 = {}, acc11 = {};
  for (int kt = 0; kt < K; kt += 32) {
    // ---- stage A tile (TM x 32): 8 consecutive floats per thread, 2 strips
    #pragma unroll
    for (int rr = 0; rr < 2; ++rr) {
      const int gm = mBase + rr * 32 + srow;
      v8bf pk = {};
      if (gm < M) {
        const float4* ap = (const float4*)(A + (long)gm * K + kt + kc);
        const float4 x0 = ap[0];
        const float4 x1 = ap[1];
        pk[0] = f2bf(x0.x); pk[1] = f2bf(x0.y); pk[2] = f2bf(x0.z); pk[3] = f2bf(x0.w);
        pk[4] = f2bf(x1.x); pk[5] = f2bf(x1.y); pk[6] = f2bf(x1.z); pk[7] = f2bf(x1.w);
        if (kt + 32 < K)
          __builtin_prefetch(A + (long)gm * K + kt + 32 + kc, 0, 1);
      }
      *(v8bf*)&As[(rr * 32 + srow) * LDT + kpos] = pk;
    }
    // ---- stage B tile (32 x TN): two 32-col halves, coalesced k-rows ----
    #pragma unroll
    for (int hh = 0; hh < 2; ++hh) {
      const int gn = nBase + hh * 32 + srow;
      v8bf pk = {};
      if (gn < N) {
        #pragma unroll
        for (int j = 0; j < 8; ++j)
          pk[j] = f2bf(Bw[(long)(kt + kc + j) * N + gn]);
        if (kt + 32 < K)
          __builtin_prefetch(Bw + (long)(kt + 32 + kc) * N + gn, 0, 1);
      }
      *(v8bf*)&Bs[(hh * 32 + srow) * LDT + kpos] = pk;
    }
    __syncthreads();

    // ---- fragments: two ds_load_b128 each ----
    union Frag { v16bf v; v8bf h[2]; };
    const int r  = lane & 15;
    const int ho = (lane >> 4) * 16;    // lanes 0-15 -> 0, lanes 16-31 -> 16
    Frag a0, a1, b0, b1;
    a0.h[0] = *(const v8bf*)&As[(mOff +      r) * LDT + ho];
    a0.h[1] = *(const v8bf*)&As[(mOff +      r) * LDT + ho + 8];
    a1.h[0] = *(const v8bf*)&As[(mOff + 16 + r) * LDT + ho];
    a1.h[1] = *(const v8bf*)&As[(mOff + 16 + r) * LDT + ho + 8];
    b0.h[0] = *(const v8bf*)&Bs[(nOff +      r) * LDT + ho];
    b0.h[1] = *(const v8bf*)&Bs[(nOff +      r) * LDT + ho + 8];
    b1.h[0] = *(const v8bf*)&Bs[(nOff + 16 + r) * LDT + ho];
    b1.h[1] = *(const v8bf*)&Bs[(nOff + 16 + r) * LDT + ho + 8];

    acc00 = __builtin_amdgcn_wmma_f32_16x16x32_bf16(
        false, a0.v, false, b0.v, (short)0, acc00, false, false);
    acc01 = __builtin_amdgcn_wmma_f32_16x16x32_bf16(
        false, a0.v, false, b1.v, (short)0, acc01, false, false);
    acc10 = __builtin_amdgcn_wmma_f32_16x16x32_bf16(
        false, a1.v, false, b0.v, (short)0, acc10, false, false);
    acc11 = __builtin_amdgcn_wmma_f32_16x16x32_bf16(
        false, a1.v, false, b1.v, (short)0, acc11, false, false);
    __syncthreads();
  }

  // ---- epilogue: C/D layout -> N = lane&15 (+tile), M = r2 (+8 for lanes>=16)
  const int n0 = nBase + nOff + (lane & 15);
  const int n1 = n0 + 16;
  const int mTop = mBase + mOff + ((lane >> 4) ? 8 : 0);
  const float bv0 = (bias != nullptr && n0 < N) ? bias[n0] : 0.0f;
  const float bv1 = (bias != nullptr && n1 < N) ? bias[n1] : 0.0f;
  const v8f* accRow0[2] = {&acc00, &acc01};
  const v8f* accRow1[2] = {&acc10, &acc11};
  #pragma unroll
  for (int mi = 0; mi < 2; ++mi) {
    const v8f* const* ar = (mi == 0) ? accRow0 : accRow1;
    const int m0 = mTop + mi * 16;
    #pragma unroll
    for (int r2 = 0; r2 < 8; ++r2) {
      const int m = m0 + r2;
      if (m < M) {
        if (n0 < N) {
          float v = (*ar[0])[r2] + bv0;
          if (act) v = fmaxf(v, 0.0f);
          C[(long)m * N + n0] = v;
        }
        if (n1 < N) {
          float v = (*ar[1])[r2] + bv1;
          if (act) v = fmaxf(v, 0.0f);
          C[(long)m * N + n1] = v;
        }
      }
    }
  }
}

// ---------------------------------------------------------------------------
// Online-softmax attention: one thread per (b,h,q) row; hd=32, float4 loads.
// qh: [B*Q, D]; kh/vh: [B*Nk, D]; bias: [B*Q, Nk] additive or null.
// ---------------------------------------------------------------------------
__global__ void attn_kernel(const float* __restrict__ qh, const float* __restrict__ kh,
                            const float* __restrict__ vh, const float* __restrict__ bias,
                            float* __restrict__ out, int Nk) {
  const int idx = blockIdx.x * blockDim.x + threadIdx.x;
  if (idx >= B_ * H_ * Q_) return;
  const int q = idx % Q_;
  const int h = (idx / Q_) % H_;
  const int b = idx / (Q_ * H_);
  const float scale = 0.17677669529663687f;  // 1/sqrt(32)

  float4 qr[HD_ / 4];
  const float4* qp = (const float4*)(qh + ((long)(b * Q_ + q) * D_ + h * HD_));
  #pragma unroll
  for (int j = 0; j < HD_ / 4; ++j) qr[j] = qp[j];

  float m = -3.0e38f, l = 0.0f;
  float4 o[HD_ / 4];
  #pragma unroll
  for (int j = 0; j < HD_ / 4; ++j) o[j] = make_float4(0.f, 0.f, 0.f, 0.f);

  const float* bp = bias ? (bias + (long)(b * Q_ + q) * Nk) : nullptr;
  for (int k = 0; k < Nk; ++k) {
    const float4* kp = (const float4*)(kh + ((long)(b * Nk + k) * D_ + h * HD_));
    float s = 0.0f;
    #pragma unroll
    for (int j = 0; j < HD_ / 4; ++j) {
      const float4 kv = kp[j];
      s += qr[j].x * kv.x + qr[j].y * kv.y + qr[j].z * kv.z + qr[j].w * kv.w;
    }
    s *= scale;
    if (bp) s += bp[k];
    const float mn   = fmaxf(m, s);
    const float corr = __expf(m - mn);
    const float p    = __expf(s - mn);
    l = l * corr + p;
    const float4* vp = (const float4*)(vh + ((long)(b * Nk + k) * D_ + h * HD_));
    #pragma unroll
    for (int j = 0; j < HD_ / 4; ++j) {
      const float4 vv = vp[j];
      o[j].x = o[j].x * corr + p * vv.x;
      o[j].y = o[j].y * corr + p * vv.y;
      o[j].z = o[j].z * corr + p * vv.z;
      o[j].w = o[j].w * corr + p * vv.w;
    }
    m = mn;
  }
  const float inv = 1.0f / l;
  float4* op = (float4*)(out + ((long)(b * Q_ + q) * D_ + h * HD_));
  #pragma unroll
  for (int j = 0; j < HD_ / 4; ++j) {
    op[j] = make_float4(o[j].x * inv, o[j].y * inv, o[j].z * inv, o[j].w * inv);
  }
}

// ---------------------------------------------------------------------------
// out[row] = LayerNorm(x[row] (+ resid[row])) * s + b ; one block per row.
// ---------------------------------------------------------------------------
__global__ __launch_bounds__(256)
void add_ln_kernel(const float* __restrict__ x, const float* __restrict__ resid,
                   const float* __restrict__ s, const float* __restrict__ bp,
                   float* __restrict__ out) {
  __shared__ float red[256];
  __shared__ float red2[256];
  const int row = blockIdx.x;
  const int d = threadIdx.x;
  float v = x[(long)row * D_ + d];
  if (resid) v += resid[(long)row * D_ + d];
  red[d] = v; red2[d] = v * v;
  __syncthreads();
  for (int off = 128; off > 0; off >>= 1) {
    if (d < off) { red[d] += red[d + off]; red2[d] += red2[d + off]; }
    __syncthreads();
  }
  const float mean = red[0] * (1.0f / D_);
  const float var  = red2[0] * (1.0f / D_) - mean * mean;
  const float r = rsqrtf(var + 1e-5f);
  out[(long)row * D_ + d] = (v - mean) * r * s[d] + bp[d];
}

// ---------------------------------------------------------------------------
// Attention bias from previous mask: bilinear resize 112x112 -> hxw,
// ignore = sigmoid(v)<0.5 (i.e. v<0); fully-ignored rows are un-masked.
// One block per (b,q).
// ---------------------------------------------------------------------------
__global__ __launch_bounds__(256)
void mask_bias_kernel(const float* __restrict__ prev, float* __restrict__ bias,
                      int h, int w) {
  __shared__ int cnt;
  const int bq = blockIdx.x;
  const int N = h * w;
  const float* pp = prev + (long)bq * (HM_ * HM_);
  float* bo = bias + (long)bq * N;
  if (threadIdx.x == 0) cnt = 0;
  __syncthreads();
  int local = 0;
  for (int i = threadIdx.x; i < N; i += blockDim.x) {
    const int oy = i / w, ox = i % w;
    const float fy = (oy + 0.5f) * ((float)HM_ / (float)h) - 0.5f;
    const float fx = (ox + 0.5f) * ((float)HM_ / (float)w) - 0.5f;
    const int y0 = (int)floorf(fy); const float wy = fy - (float)y0;
    const int x0 = (int)floorf(fx); const float wx = fx - (float)x0;
    const int y0c = min(max(y0, 0), HM_ - 1);
    const int y1c = min(max(y0 + 1, 0), HM_ - 1);
    const int x0c = min(max(x0, 0), HM_ - 1);
    const int x1c = min(max(x0 + 1, 0), HM_ - 1);
    const float v00 = pp[y0c * HM_ + x0c], v01 = pp[y0c * HM_ + x1c];
    const float v10 = pp[y1c * HM_ + x0c], v11 = pp[y1c * HM_ + x1c];
    const float v = (v00 * (1.f - wx) + v01 * wx) * (1.f - wy)
                  + (v10 * (1.f - wx) + v11 * wx) * wy;
    const int ign = (v < 0.0f);
    local += ign;
    bo[i] = ign ? -1e9f : 0.0f;
  }
  atomicAdd(&cnt, local);
  __syncthreads();
  if (cnt == N) {
    for (int i = threadIdx.x; i < N; i += blockDim.x) bo[i] = 0.0f;
  }
}

// mem [B,D,P] -> memT [B,P,D]
__global__ void transpose_kernel(const float* __restrict__ in, float* __restrict__ out, int P) {
  const long idx = (long)blockIdx.x * blockDim.x + threadIdx.x;
  const long total = (long)B_ * D_ * P;
  if (idx >= total) return;
  const int p = (int)(idx % P);
  const int d = (int)((idx / P) % D_);
  const int b = (int)(idx / ((long)P * D_));
  out[((long)b * P + p) * D_ + d] = in[idx];
}

__global__ void initq_kernel(const float* __restrict__ qf, const float* __restrict__ qe,
                             float* __restrict__ q) {
  const int idx = blockIdx.x * blockDim.x + threadIdx.x;
  if (idx >= B_ * Q_ * D_) return;
  const int qd = idx % (Q_ * D_);
  q[idx] = qf[qd] + qe[qd];
}

// ---------------------------------------------------------------------------
static inline void gemm(hipStream_t st, const float* A, const float* Bw, float* C,
                        const float* bias, int M, int N, int K,
                        long sA, long sB, long sC, int batch, int act) {
  dim3 grid((N + TN - 1) / TN, (M + TM - 1) / TM, batch);
  gemm_bf16_wmma<<<grid, 128, 0, st>>>(A, Bw, C, bias, M, N, K, sA, sB, sC, act);
}

extern "C" void kernel_launch(void* const* d_in, const int* in_sizes, int n_in,
                              void* d_out, int out_size, void* d_ws, size_t ws_size,
                              hipStream_t stream) {
  (void)in_sizes; (void)n_in; (void)out_size; (void)ws_size;
  const float* mask_features = (const float*)d_in[0];
  const float* memi[3] = {(const float*)d_in[1], (const float*)d_in[2], (const float*)d_in[3]};
  const float* query_feat  = (const float*)d_in[4];
  const float* query_embed = (const float*)d_in[5];
  const float* ca_wqkv = (const float*)d_in[6];
  const float* ca_bqkv = (const float*)d_in[7];
  const float* ca_wo   = (const float*)d_in[8];
  const float* ca_bo   = (const float*)d_in[9];
  const float* ca_ln_s = (const float*)d_in[10];
  const float* ca_ln_b = (const float*)d_in[11];
  const float* sa_wqkv = (const float*)d_in[12];
  const float* sa_bqkv = (const float*)d_in[13];
  const float* sa_wo   = (const float*)d_in[14];
  const float* sa_bo   = (const float*)d_in[15];
  const float* sa_ln_s = (const float*)d_in[16];
  const float* sa_ln_b = (const float*)d_in[17];
  const float* ffn_w1  = (const float*)d_in[18];
  const float* ffn_b1  = (const float*)d_in[19];
  const float* ffn_w2  = (const float*)d_in[20];
  const float* ffn_b2  = (const float*)d_in[21];
  const float* ffn_ln_s = (const float*)d_in[22];
  const float* ffn_ln_b = (const float*)d_in[23];
  const float* dec_ln_s = (const float*)d_in[24];
  const float* dec_ln_b = (const float*)d_in[25];
  const float* me_w1 = (const float*)d_in[26];
  const float* me_b1 = (const float*)d_in[27];
  const float* me_w2 = (const float*)d_in[28];
  const float* me_b2 = (const float*)d_in[29];
  const float* me_w3 = (const float*)d_in[30];
  const float* me_b3 = (const float*)d_in[31];
  const float* cls_w = (const float*)d_in[32];
  const float* cls_b = (const float*)d_in[33];
  float* out = (float*)d_out;

  // ---- workspace carve ----
  size_t off = 0;
  auto alloc = [&](size_t elems) {
    float* p = (float*)((char*)d_ws + off);
    off += ((elems * sizeof(float) + 255) / 256) * 256;
    return p;
  };
  const int Pm[3] = {14 * 14, 28 * 28, 56 * 56};
  const int HWs[3] = {14, 28, 56};
  float* memT[3];
  for (int j = 0; j < 3; ++j) memT[j] = alloc((size_t)B_ * Pm[j] * D_);
  const int RQ = B_ * Q_;                     // 800 rows
  float* qbuf    = alloc((size_t)RQ * D_);
  float* qh      = alloc((size_t)RQ * D_);
  float* kh      = alloc((size_t)B_ * 3136 * D_);
  float* vh      = alloc((size_t)B_ * 3136 * D_);
  float* attnout = alloc((size_t)RQ * D_);
  float* proj    = alloc((size_t)RQ * D_);
  float* ffnh    = alloc((size_t)RQ * F_);
  float* nq      = alloc((size_t)RQ * D_);
  float* me1     = alloc((size_t)RQ * D_);
  float* me2     = alloc((size_t)RQ * D_);
  float* pmask   = alloc((size_t)B_ * Q_ * (HM_ * HM_));
  float* biasb   = alloc((size_t)B_ * Q_ * 3136);

  // ---- preprocessing ----
  for (int j = 0; j < 3; ++j) {
    const long tot = (long)B_ * D_ * Pm[j];
    transpose_kernel<<<(int)((tot + 255) / 256), 256, 0, stream>>>(memi[j], memT[j], Pm[j]);
  }
  initq_kernel<<<(B_ * Q_ * D_ + 255) / 256, 256, 0, stream>>>(query_feat, query_embed, qbuf);

  const int attnBlocks = (B_ * H_ * Q_ + 127) / 128;
  const float* prev = nullptr;

  for (int i = 0; i < L_; ++i) {
    const int mi = i % 3;
    const int Nk = Pm[mi];
    const int hw = HWs[mi];

    // ---- cross-attention ----
    const float* W  = ca_wqkv + (size_t)(i * 3) * D_ * D_;
    const float* Bq = ca_bqkv + (size_t)(i * 3) * D_;
    gemm(stream, qbuf,     W,                qh, Bq,          RQ,       D_, D_, 0, 0, 0, 1, 0);
    gemm(stream, memT[mi], W + D_ * D_,      kh, Bq + D_,     B_ * Nk,  D_, D_, 0, 0, 0, 1, 0);
    gemm(stream, memT[mi], W + 2 * D_ * D_,  vh, Bq + 2 * D_, B_ * Nk,  D_, D_, 0, 0, 0, 1, 0);
    const float* bptr = nullptr;
    if (prev) {
      mask_bias_kernel<<<RQ, 256, 0, stream>>>(prev, biasb, hw, hw);
      bptr = biasb;
    }
    attn_kernel<<<attnBlocks, 128, 0, stream>>>(qh, kh, vh, bptr, attnout, Nk);
    gemm(stream, attnout, ca_wo + (size_t)i * D_ * D_, proj, ca_bo + (size_t)i * D_,
         RQ, D_, D_, 0, 0, 0, 1, 0);
    add_ln_kernel<<<RQ, 256, 0, stream>>>(proj, qbuf, ca_ln_s + i * D_, ca_ln_b + i * D_, qbuf);

    // ---- self-attention ----
    const float* Ws  = sa_wqkv + (size_t)(i * 3) * D_ * D_;
    const float* Bsb = sa_bqkv + (size_t)(i * 3) * D_;
    gemm(stream, qbuf, Ws,                qh, Bsb,          RQ, D_, D_, 0, 0, 0, 1, 0);
    gemm(stream, qbuf, Ws + D_ * D_,      kh, Bsb + D_,     RQ, D_, D_, 0, 0, 0, 1, 0);
    gemm(stream, qbuf, Ws + 2 * D_ * D_,  vh, Bsb + 2 * D_, RQ, D_, D_, 0, 0, 0, 1, 0);
    attn_kernel<<<attnBlocks, 128, 0, stream>>>(qh, kh, vh, nullptr, attnout, Q_);
    gemm(stream, attnout, sa_wo + (size_t)i * D_ * D_, proj, sa_bo + (size_t)i * D_,
         RQ, D_, D_, 0, 0, 0, 1, 0);
    add_ln_kernel<<<RQ, 256, 0, stream>>>(proj, qbuf, sa_ln_s + i * D_, sa_ln_b + i * D_, qbuf);

    // ---- FFN ----
    gemm(stream, qbuf, ffn_w1 + (size_t)i * D_ * F_, ffnh, ffn_b1 + (size_t)i * F_,
         RQ, F_, D_, 0, 0, 0, 1, 1);
    gemm(stream, ffnh, ffn_w2 + (size_t)i * F_ * D_, proj, ffn_b2 + (size_t)i * D_,
         RQ, D_, F_, 0, 0, 0, 1, 0);
    add_ln_kernel<<<RQ, 256, 0, stream>>>(proj, qbuf, ffn_ln_s + i * D_, ffn_ln_b + i * D_, qbuf);

    // ---- decoder norm + mask embed MLP ----
    add_ln_kernel<<<RQ, 256, 0, stream>>>(qbuf, nullptr, dec_ln_s, dec_ln_b, nq);
    gemm(stream, nq,  me_w1, me1, me_b1, RQ, D_, D_, 0, 0, 0, 1, 1);
    gemm(stream, me1, me_w2, me2, me_b2, RQ, D_, D_, 0, 0, 0, 1, 1);
    gemm(stream, me2, me_w3, me1, me_b3, RQ, D_, D_, 0, 0, 0, 1, 0);

    // ---- pred_mask: batched [Q,D] @ [D, Hm*Wm] (mask_features is [B,D,P] row-major)
    float* pm = (i == L_ - 1) ? (out + (size_t)B_ * Q_ * C_) : pmask;
    gemm(stream, me1, mask_features, pm, nullptr,
         Q_, HM_ * HM_, D_,
         (long)Q_ * D_, (long)D_ * HM_ * HM_, (long)Q_ * HM_ * HM_, B_, 0);
    prev = pm;
  }

  // ---- final classifier: pred_logits = nq @ cls_w + cls_b ----
  gemm(stream, nq, cls_w, out, cls_b, RQ, C_, D_, 0, 0, 0, 1, 0);
}
